// Model_86277303042316
// MI455X (gfx1250) — matrix operand order repaired and checked by reading, not compiled, not run on hardware
//
#include <hip/hip_runtime.h>
#include <hip/hip_bf16.h>
#include <stdint.h>

// ---------------------------------------------------------------------------
// BiLSTM autoencoder for MI455X (gfx1250, wave32, WMMA).
//  - bf16 WMMA 16x16x32, fp32 accumulate
//  - input projections: full-GPU GEMMs (Zin = x@Wi + b), output stored bf16 in
//    FRAGMENT-PACKED tiles (tile*256 + lane*8) -> b128 stores/loads
//  - recurrent scan: 16 workgroups (8 batch-chunks x 2 dirs), h in LDS (bf16),
//    c in VGPRs, Zin tile loaded as the WMMA C-accumulator init (1 b128/lane),
//    next-timestep Zin prefetched with global_prefetch_b8.
//  - A fragments loaded as contiguous b128 runs (2 per fragment).
// ---------------------------------------------------------------------------

#define DEVINL __device__ __forceinline__

typedef __attribute__((ext_vector_type(16))) __bf16 bf16x16;
typedef __attribute__((ext_vector_type(2)))  __bf16 bf16x2;
typedef __attribute__((ext_vector_type(8)))  float  f32x8;

union AFrag { bf16x16 v; uint4 q[2]; uint32_t u32[8]; uint16_t u16[16]; };
union BFrag { bf16x16 v; uint4 q[2]; uint32_t u32[8]; uint16_t u16[16]; };
union CFrag { f32x8  v; float f[8]; };

DEVINL uint16_t f2bf(float f) {
  uint32_t u = __float_as_uint(f);
  uint32_t r = u + 0x7FFFu + ((u >> 16) & 1u);   // round-to-nearest-even
  return (uint16_t)(r >> 16);
}
DEVINL float bf2f(uint16_t s) { return __uint_as_float(((uint32_t)s) << 16); }

// packed f32x2 -> bf16x2 (low = a). HW packed converter when available.
DEVINL uint32_t f2bf_pk(float a, float b) {
#if __has_builtin(__builtin_amdgcn_cvt_pk_bf16_f32)
  union { bf16x2 v; uint32_t u; } cv;
  cv.v = __builtin_amdgcn_cvt_pk_bf16_f32(a, b);
  return cv.u;
#else
  uint32_t ua = __float_as_uint(a), ub = __float_as_uint(b);
  uint32_t ra = ua + 0x7FFFu + ((ua >> 16) & 1u);
  uint32_t rb = ub + 0x7FFFu + ((ub >> 16) & 1u);
  return __builtin_amdgcn_perm(rb, ra, 0x07060302u);  // {rb[31:16], ra[31:16]}
#endif
}
DEVINL float bf2f_lo(uint32_t w) { return __uint_as_float(w << 16); }
DEVINL float bf2f_hi(uint32_t w) { return __uint_as_float(w & 0xFFFF0000u); }

DEVINL float sigf(float x)  { return 1.0f / (1.0f + __expf(-x)); }
DEVINL float reluf(float x) { return fmaxf(x, 0.0f); }

// Problem constants
static constexpr int Bb = 128;   // batch
static constexpr int Tt = 256;   // timesteps

// ---------------------------------------------------------------------------
// Pack weight W[K][N] (fp32, row-major, per-dir contiguous) into bf16
// B-fragment tiles: tile (nt,kt) = 32(K) x 16(N); element (lane, i) holds
// W[kt*32 + 16*(lane/16) + i, nt*16 + lane%16].
// Packed index: ((nt*KT + kt) * 512) + lane*16 + i.
// ---------------------------------------------------------------------------
__global__ void pack_b_kernel(const float* __restrict__ W, uint16_t* __restrict__ out,
                              int K, int N) {
  const int total = K * N;
  const int p = blockIdx.x * blockDim.x + threadIdx.x;
  const int dir = blockIdx.y;
  if (p >= total) return;
  const float* Wd = W + (size_t)dir * total;
  uint16_t*    Od = out + (size_t)dir * total;
  const int tile = p >> 9, q = p & 511;
  const int lane = q >> 4, i = q & 15;
  const int KT = K >> 5;
  const int kt = tile % KT, nt = tile / KT;
  const int k = kt * 32 + ((lane >> 4) << 4) + i;
  const int n = nt * 16 + (lane & 15);
  Od[p] = f2bf(Wd[(size_t)k * N + n]);
}

// ---------------------------------------------------------------------------
// GEMM: Z[r][col] = A[row r][0..K) @ Wpacked + bias[col]
// rows r map to (t = r / Bsz, b = r % Bsz); A row addr = t*sAt + b*sAb.
// mode 0: store bf16, fragment-packed: Zout[dir*zds + (rt*NTC+ct)*256 + lane*8]
// mode 1: store f32 to Fout[(b*T + t)*cols + col]   (final dense layer)
// Block: 256 thr = 8 waves; wave = one 16-row tile x 64 cols (4 N-tiles).
// Requires R % 128 == 0, K % 32 == 0, cols % 64 == 0.
// ---------------------------------------------------------------------------
__global__ __launch_bounds__(256) void proj_gemm_kernel(
    const float* __restrict__ A, const uint16_t* __restrict__ Bp,
    const float* __restrict__ bias,
    uint16_t* __restrict__ Zout, float* __restrict__ Fout,
    int Bsz, int K, int cols, int sAt, int sAb,
    size_t bp_dir_stride, int bias_dir_stride, size_t z_dir_stride,
    int Tdim, int mode)
{
  const int wave = threadIdx.x >> 5;
  const int lane = threadIdx.x & 31;
  const int half = lane >> 4;
  const int lm   = lane & 15;
  const int dir  = blockIdx.z;
  const int rt   = blockIdx.x * 8 + wave;     // row-tile index
  const int r0   = rt * 16;
  const int col0 = blockIdx.y * 64;
  const int KT   = K >> 5;
  const int NTC  = cols >> 4;

  const uint16_t* Bd    = Bp + (size_t)dir * bp_dir_stride;
  const float*    biasd = bias + (size_t)dir * bias_dir_stride;

  // A-fragment row for this lane (both lane halves carry M = lane%16)
  const int gr = r0 + lm;
  const int tA = gr / Bsz, bA = gr % Bsz;
  const float* Arow = A + (size_t)tA * sAt + (size_t)bA * sAb;

  CFrag acc[4];
  #pragma unroll
  for (int ct = 0; ct < 4; ++ct)
    #pragma unroll
    for (int v = 0; v < 8; ++v) acc[ct].f[v] = 0.0f;

  for (int kk = 0; kk < KT; ++kk) {
    // A fragment: element i <-> K = kk*32 + 16*half + i
    // -> one contiguous 16-float run at k = kk*32 + 16*half
    const float* ap = Arow + kk * 32 + 16 * half;
    const float4 fa = *(const float4*)(ap);
    const float4 fb = *(const float4*)(ap + 4);
    const float4 fc = *(const float4*)(ap + 8);
    const float4 fd = *(const float4*)(ap + 12);
    AFrag af;
    af.u32[0] = f2bf_pk(fa.x, fa.y);
    af.u32[1] = f2bf_pk(fa.z, fa.w);
    af.u32[2] = f2bf_pk(fb.x, fb.y);
    af.u32[3] = f2bf_pk(fb.z, fb.w);
    af.u32[4] = f2bf_pk(fc.x, fc.y);
    af.u32[5] = f2bf_pk(fc.z, fc.w);
    af.u32[6] = f2bf_pk(fd.x, fd.y);
    af.u32[7] = f2bf_pk(fd.z, fd.w);

    #pragma unroll
    for (int ct = 0; ct < 4; ++ct) {
      const int nt = (col0 >> 4) + ct;
      const uint16_t* tadr = Bd + (((size_t)nt * KT + kk) << 9) + (lane << 4);
      if (kk + 1 < KT)
        __builtin_prefetch(tadr + 512, 0, 1);   // global_prefetch_b8
      BFrag bf;
      const uint4* p = (const uint4*)tadr;
      bf.q[0] = p[0]; bf.q[1] = p[1];
      acc[ct].v = __builtin_amdgcn_wmma_f32_16x16x32_bf16(
          false, af.v, false, bf.v, (short)0, acc[ct].v, false, false);
    }
  }

  // D fragment: (m = v + 8*half, n = lane%16)
  #pragma unroll
  for (int ct = 0; ct < 4; ++ct) {
    const int col = col0 + ct * 16 + lm;
    const float bv = biasd[col];
    if (mode == 0) {
      // fragment-packed bf16 tile, one b128 store per lane
      uint4 st;
      st.x = f2bf_pk(acc[ct].f[0] + bv, acc[ct].f[1] + bv);
      st.y = f2bf_pk(acc[ct].f[2] + bv, acc[ct].f[3] + bv);
      st.z = f2bf_pk(acc[ct].f[4] + bv, acc[ct].f[5] + bv);
      st.w = f2bf_pk(acc[ct].f[6] + bv, acc[ct].f[7] + bv);
      const int ctg = (col0 >> 4) + ct;
      uint16_t* dst = Zout + (size_t)dir * z_dir_stride
                    + (((size_t)rt * NTC + ctg) << 8) + (lane << 3);
      *(uint4*)dst = st;
    } else {
      #pragma unroll
      for (int v = 0; v < 8; ++v) {
        const int m = v + 8 * half;
        const int grr = r0 + m;
        const int t2 = grr / Bsz, b2 = grr % Bsz;
        Fout[((size_t)b2 * Tdim + t2) * cols + col] = acc[ct].f[v] + bv;
      }
    }
  }
}

// ---------------------------------------------------------------------------
// Recurrent scan. One workgroup = 16 batch rows x one direction, loops over T.
// z = Zin[t] + h_prev @ Wh  (WMMA, C init from fragment-packed Zin);
// Keras gates i,f,g,o:  c = sig(f)*c + sig(i)*relu(g);  h = sig(o)*relu(c)
// h kept bf16 in LDS (A-matrix source, read as 2x ds_load_b128 per fragment);
// c kept in VGPRs.  Wave w owns hidden columns [w*16*TG, (w+1)*16*TG) of every
// gate, so the i/f/g/o values for a hidden unit live in the same lane.
// ---------------------------------------------------------------------------
template <int U>
__global__ __launch_bounds__(256) void lstm_scan_kernel(
    const uint16_t* __restrict__ Zin, size_t z_dir_stride, size_t z_t_stride,
    const uint16_t* __restrict__ Whp, size_t wh_dir_stride,
    float* __restrict__ outSeq, float* __restrict__ outFin,
    int T, int Bsz, int seq)
{
  constexpr int TG  = U / 128;   // N-tiles per gate per wave
  constexpr int KT  = U / 32;    // K steps
  constexpr int U16 = U / 16;    // N-tiles per gate
  constexpr int NTC = 4 * U16;   // N-tiles total (= cols/16)
  __shared__ uint16_t hbuf[16 * U];

  const int wave = threadIdx.x >> 5;
  const int lane = threadIdx.x & 31;
  const int half = lane >> 4;
  const int lm   = lane & 15;
  const int dir  = blockIdx.y;
  const int rtl  = blockIdx.x;       // batch row-tile (16 rows)
  const int b0   = rtl * 16;

  const uint16_t* Zd = Zin + (size_t)dir * z_dir_stride;
  const uint16_t* Wd = Whp + (size_t)dir * wh_dir_stride;

  for (int i = threadIdx.x; i < 16 * U; i += 256) hbuf[i] = 0;  // h0 = 0
  float creg[TG][8];
  float hl[TG][8];
  #pragma unroll
  for (int tt = 0; tt < TG; ++tt)
    #pragma unroll
    for (int v = 0; v < 8; ++v) { creg[tt][v] = 0.0f; hl[tt][v] = 0.0f; }
  __syncthreads();

  for (int t = 0; t < T; ++t) {
    const int te = dir ? (T - 1 - t) : t;
    const size_t zbase = (size_t)te * z_t_stride + ((size_t)rtl * NTC << 8);

    // prefetch next timestep's Zin slice (contiguous NTC*512 bytes)
    if (t + 1 < T) {
      const int tn = dir ? (T - 2 - t) : (t + 1);
      const size_t znext = (size_t)tn * z_t_stride + ((size_t)rtl * NTC << 8);
      const char* pf = (const char*)(Zd + znext) + threadIdx.x * (NTC * 2);
      __builtin_prefetch(pf, 0, 0);   // global_prefetch_b8
    }

    // C init = gate pre-activations (xWi + b): one b128 load per tile per lane
    CFrag acc[4][TG];
    #pragma unroll
    for (int g = 0; g < 4; ++g)
      #pragma unroll
      for (int tt = 0; tt < TG; ++tt) {
        const int nt = g * U16 + wave * TG + tt;
        const uint4 q = *(const uint4*)(Zd + zbase + ((size_t)nt << 8) + (lane << 3));
        acc[g][tt].f[0] = bf2f_lo(q.x); acc[g][tt].f[1] = bf2f_hi(q.x);
        acc[g][tt].f[2] = bf2f_lo(q.y); acc[g][tt].f[3] = bf2f_hi(q.y);
        acc[g][tt].f[4] = bf2f_lo(q.z); acc[g][tt].f[5] = bf2f_hi(q.z);
        acc[g][tt].f[6] = bf2f_lo(q.w); acc[g][tt].f[7] = bf2f_hi(q.w);
      }

    // z += h_prev @ Wh
    for (int kk = 0; kk < KT; ++kk) {
      // A fragment from LDS: two contiguous 16B runs per lane
      AFrag af;
      const uint16_t* hrow = &hbuf[lm * U + kk * 32 + 16 * half];
      af.q[0] = *(const uint4*)(hrow);
      af.q[1] = *(const uint4*)(hrow + 8);
      #pragma unroll
      for (int g = 0; g < 4; ++g)
        #pragma unroll
        for (int tt = 0; tt < TG; ++tt) {
          const int nt = g * U16 + wave * TG + tt;
          const uint16_t* tadr = Wd + (((size_t)nt * KT + kk) << 9) + (lane << 4);
          BFrag bf;
          const uint4* p = (const uint4*)tadr;
          bf.q[0] = p[0]; bf.q[1] = p[1];
          acc[g][tt].v = __builtin_amdgcn_wmma_f32_16x16x32_bf16(
              false, af.v, false, bf.v, (short)0, acc[g][tt].v, false, false);
        }
    }

    // gates (Keras LSTM, activation=relu, recurrent_activation=sigmoid)
    #pragma unroll
    for (int tt = 0; tt < TG; ++tt)
      #pragma unroll
      for (int v = 0; v < 8; ++v) {
        const float zi = acc[0][tt].f[v], zf = acc[1][tt].f[v];
        const float zg = acc[2][tt].f[v], zo = acc[3][tt].f[v];
        const float cc = sigf(zf) * creg[tt][v] + sigf(zi) * reluf(zg);
        creg[tt][v] = cc;
        hl[tt][v] = sigf(zo) * reluf(cc);
      }

    __syncthreads();   // all waves done reading hbuf this step
    #pragma unroll
    for (int tt = 0; tt < TG; ++tt) {
      const int j = (wave * TG + tt) * 16 + lm;
      #pragma unroll
      for (int v = 0; v < 8; ++v) {
        const int m = v + 8 * half;
        hbuf[m * U + j] = f2bf(hl[tt][v]);
        if (seq)
          outSeq[(size_t)te * Bsz * (2 * U) + (size_t)(b0 + m) * (2 * U)
                 + dir * U + j] = hl[tt][v];
      }
    }
    __syncthreads();   // hbuf ready for next step
  }

  if (!seq) {  // enc2: final hidden state only -> [B, 2U]
    #pragma unroll
    for (int tt = 0; tt < TG; ++tt) {
      const int j = (wave * TG + tt) * 16 + lm;
      #pragma unroll
      for (int v = 0; v < 8; ++v) {
        const int m = v + 8 * half;
        outFin[(size_t)(b0 + m) * (2 * U) + dir * U + j] = hl[tt][v];
      }
    }
  }
}

// ---------------------------------------------------------------------------
extern "C" void kernel_launch(void* const* d_in, const int* in_sizes, int n_in,
                              void* d_out, int out_size, void* d_ws, size_t ws_size,
                              hipStream_t stream) {
  const float* x    = (const float*)d_in[0];
  const float* e1Wi = (const float*)d_in[1];
  const float* e1Wh = (const float*)d_in[2];
  const float* e1b  = (const float*)d_in[3];
  const float* e2Wi = (const float*)d_in[4];
  const float* e2Wh = (const float*)d_in[5];
  const float* e2b  = (const float*)d_in[6];
  const float* d1Wi = (const float*)d_in[7];
  const float* d1Wh = (const float*)d_in[8];
  const float* d1b  = (const float*)d_in[9];
  const float* d2Wi = (const float*)d_in[10];
  const float* d2Wh = (const float*)d_in[11];
  const float* d2b  = (const float*)d_in[12];
  const float* Wd   = (const float*)d_in[13];
  const float* bd   = (const float*)d_in[14];
  float* out = (float*)d_out;

  char* ws = (char*)d_ws;
  size_t off = 0;
  auto take = [&](size_t bytes) -> char* {
    char* p = ws + off;
    off += (bytes + 255) & ~(size_t)255;
    return p;
  };

  // packed bf16 weights (per-dir contiguous)
  uint16_t* p_e1Wi = (uint16_t*)take(2ull *  64 * 1024 * 2);
  uint16_t* p_e1Wh = (uint16_t*)take(2ull * 256 * 1024 * 2);
  uint16_t* p_e2Wi = (uint16_t*)take(2ull * 512 *  512 * 2);
  uint16_t* p_e2Wh = (uint16_t*)take(2ull * 128 *  512 * 2);
  uint16_t* p_d1Wi = (uint16_t*)take(2ull * 256 *  512 * 2);
  uint16_t* p_d1Wh = (uint16_t*)take(2ull * 128 *  512 * 2);
  uint16_t* p_d2Wi = (uint16_t*)take(2ull * 256 * 1024 * 2);
  uint16_t* p_d2Wh = (uint16_t*)take(2ull * 256 * 1024 * 2);
  uint16_t* p_Wd   = (uint16_t*)take(1ull * 512 *   64 * 2);
  // activations / gate pre-activations
  uint16_t* Zin  = (uint16_t*)take(2ull * Tt * Bb * 1024 * 2);  // shared per layer
  float*    act1 = (float*)take((size_t)Tt * Bb * 512 * 4);
  float*    zvec = (float*)take((size_t)Bb * 256 * 4);
  float*    act3 = (float*)take((size_t)Tt * Bb * 256 * 4);
  float*    act4 = (float*)take((size_t)Tt * Bb * 512 * 4);

  auto pack = [&](const float* W, uint16_t* dst, int K, int N, int dirs) {
    dim3 g((K * N + 255) / 256, dirs);
    pack_b_kernel<<<g, 256, 0, stream>>>(W, dst, K, N);
  };
  pack(e1Wi, p_e1Wi,  64, 1024, 2);
  pack(e1Wh, p_e1Wh, 256, 1024, 2);
  pack(e2Wi, p_e2Wi, 512,  512, 2);
  pack(e2Wh, p_e2Wh, 128,  512, 2);
  pack(d1Wi, p_d1Wi, 256,  512, 2);
  pack(d1Wh, p_d1Wh, 128,  512, 2);
  pack(d2Wi, p_d2Wi, 256, 1024, 2);
  pack(d2Wh, p_d2Wh, 256, 1024, 2);
  pack(Wd,   p_Wd,   512,   64, 1);

  const int R = Tt * Bb;  // 32768

  // ---- enc1: in=64, u=256, return_sequences ----
  {
    const size_t zds = (size_t)Tt * Bb * 1024;
    proj_gemm_kernel<<<dim3(R / 128, 1024 / 64, 2), 256, 0, stream>>>(
        x, p_e1Wi, e1b, Zin, nullptr, Bb, 64, 1024,
        /*sAt=*/64, /*sAb=*/Tt * 64, /*bpds=*/64ull * 1024, /*bds=*/1024,
        zds, Tt, 0);
    lstm_scan_kernel<256><<<dim3(Bb / 16, 2), 256, 0, stream>>>(
        Zin, zds, (size_t)Bb * 1024, p_e1Wh, 256ull * 1024,
        act1, nullptr, Tt, Bb, 1);
  }
  // ---- enc2: in=512, u=128, final state only ----
  {
    const size_t zds = (size_t)Tt * Bb * 512;
    proj_gemm_kernel<<<dim3(R / 128, 512 / 64, 2), 256, 0, stream>>>(
        act1, p_e2Wi, e2b, Zin, nullptr, Bb, 512, 512,
        /*sAt=*/Bb * 512, /*sAb=*/512, /*bpds=*/512ull * 512, /*bds=*/512,
        zds, Tt, 0);
    lstm_scan_kernel<128><<<dim3(Bb / 16, 2), 256, 0, stream>>>(
        Zin, zds, (size_t)Bb * 512, p_e2Wh, 128ull * 512,
        nullptr, zvec, Tt, Bb, 0);
  }
  // ---- dec1: in=256 (RepeatVector -> time-invariant Zin), u=128 ----
  {
    const size_t zds = (size_t)Bb * 512;
    proj_gemm_kernel<<<dim3(Bb / 128, 512 / 64, 2), 256, 0, stream>>>(
        zvec, p_d1Wi, d1b, Zin, nullptr, Bb, 256, 512,
        /*sAt=*/0, /*sAb=*/256, /*bpds=*/256ull * 512, /*bds=*/512,
        zds, Tt, 0);
    lstm_scan_kernel<128><<<dim3(Bb / 16, 2), 256, 0, stream>>>(
        Zin, zds, /*z_t_stride=*/0, p_d1Wh, 128ull * 512,
        act3, nullptr, Tt, Bb, 1);
  }
  // ---- dec2: in=256, u=256, return_sequences ----
  {
    const size_t zds = (size_t)Tt * Bb * 1024;
    proj_gemm_kernel<<<dim3(R / 128, 1024 / 64, 2), 256, 0, stream>>>(
        act3, p_d2Wi, d2b, Zin, nullptr, Bb, 256, 1024,
        /*sAt=*/Bb * 256, /*sAb=*/256, /*bpds=*/256ull * 1024, /*bds=*/1024,
        zds, Tt, 0);
    lstm_scan_kernel<256><<<dim3(Bb / 16, 2), 256, 0, stream>>>(
        Zin, zds, (size_t)Bb * 1024, p_d2Wh, 256ull * 1024,
        act4, nullptr, Tt, Bb, 1);
  }
  // ---- dense: [T*B,512] @ [512,64] + bd -> out[B,T,64] ----
  proj_gemm_kernel<<<dim3(R / 128, 64 / 64, 1), 256, 0, stream>>>(
      act4, p_Wd, bd, nullptr, out, Bb, 512, 64,
      /*sAt=*/Bb * 512, /*sAb=*/512, /*bpds=*/0, /*bds=*/0,
      /*zds=*/0, Tt, 1);
}
